// Attention_901943132239
// MI455X (gfx1250) — matrix operand order repaired and checked by reading, not compiled, not run on hardware
//
#include <hip/hip_runtime.h>

// Attention: scores = Q K^T, mask, softmax, O = P V ; returns (O, P).
// B=8, T=S=2048, H=64. One workgroup (8 waves, 256 thr) = one batch x one
// 16-row tile of T, covering all S=2048 columns. Scores live in LDS (fp32,
// padded rows), probabilities cached as f16 (padded rows) for the PV WMMA.

#define NB   8
#define NT   2048
#define NS   2048
#define NH   64
#define NEG_INF_F (-1.0e9f)

// Padded LDS row strides (break the stride%64==0 bank pathology).
#define SCS  2052   // score row stride in f32 words  (2052 % 64 == 4)
#define PBS  2056   // prob  row stride in f16 halves (row byte stride 4112 = 16*257)

#if __has_builtin(__builtin_amdgcn_global_load_async_to_lds_b128) && \
    __has_builtin(__builtin_amdgcn_s_wait_asynccnt)
#define USE_ASYNC_LDS 1
#else
#define USE_ASYNC_LDS 0
#endif

typedef __attribute__((ext_vector_type(16))) _Float16 v16h;
typedef __attribute__((ext_vector_type(8)))  _Float16 v8h;
typedef __attribute__((ext_vector_type(8)))  float    v8f;
typedef __attribute__((ext_vector_type(4)))  int      v4i;

#if USE_ASYNC_LDS
typedef __attribute__((address_space(1))) v4i* gv4i_ptr;   // global
typedef __attribute__((address_space(3))) v4i* lv4i_ptr;   // LDS
#endif

// ---- WMMA fragment builders (contiguous k order: k = 16*(L>>4) + e in A and B) ----

// A-matrix 16x32 f16. Lane L: row = rowbase+(L&15), kb = 16*(L>>4).
// elems 0..7  = src[kc + kb + i] ; elems 8..15 = src[kc + kb + 8 + i]
__device__ __forceinline__ v16h a_frag_from_f32(const float* __restrict__ row_ptr,
                                                int kc, int kb) {
  v16h a;
#pragma unroll
  for (int i = 0; i < 8; ++i) {
    a[i]     = (_Float16)row_ptr[kc + kb + i];
    a[i + 8] = (_Float16)row_ptr[kc + kb + 8 + i];
  }
  return a;
}

// B-matrix 32x16 f16, element e = row (kbase+e) of B at this lane's column;
// contiguous source (K^T: column n of B = row s0c+n of keys, head dim contiguous).
__device__ __forceinline__ v16h b_frag_contig_f32(const float* __restrict__ p) {
  v16h b;
#pragma unroll
  for (int i = 0; i < 16; ++i) b[i] = (_Float16)p[i];
  return b;
}

// B-matrix 32x16 f16, element e strided (V: consecutive S rows, stride NH).
__device__ __forceinline__ v16h b_frag_strided_f32(const float* __restrict__ p,
                                                   int stride) {
  v16h b;
#pragma unroll
  for (int i = 0; i < 16; ++i) b[i] = (_Float16)p[i * stride];
  return b;
}

__global__ __launch_bounds__(256) void attn_wmma_kernel(
    const float* __restrict__ q, const float* __restrict__ k,
    const float* __restrict__ v, const int* __restrict__ mask,
    float* __restrict__ o_out, float* __restrict__ p_out) {
  // Dynamic LDS carve-up (210048 B total; WGP LDS pool is 320KB).
  extern __shared__ char smem[];
  float*    sc   = (float*)(smem);               // [16][SCS] f32 scores -> exp
  _Float16* pb   = (_Float16*)(smem + 131328);   // [16][PBS] f16 probabilities
  float*    obuf = (float*)(smem + 197120);      // [16][64]  O reduction
  int*      mbuf = (int*)(smem + 201216);        // [2048] mask row
  float*    wmax = (float*)(smem + 209408);      // [8][16] per-wave row maxima
  float*    rmax = (float*)(smem + 209920);      // [16] row max
  float*    rsum = (float*)(smem + 209984);      // [16] row sum -> 1/sum

  const int tid  = threadIdx.x;
  const int lane = tid & 31;
  const int wave = tid >> 5;     // 8 waves
  const int ln16 = lane & 15;
  const int hi   = lane >> 4;    // half-wave select

  const int bidx = blockIdx.x >> 7;          // batch (128 row-tiles per batch)
  const int t0   = (blockIdx.x & 127) << 4;  // first of 16 query rows

  // ---- Stage mask row into LDS (async-to-LDS path when available) ----
#if USE_ASYNC_LDS
  {
    int* gsrc = const_cast<int*>(mask) + (size_t)bidx * NS;
#pragma unroll
    for (int it = 0; it < 2; ++it) {
      const int eidx = it * 1024 + tid * 4;  // 256 thr x b128 x 2 = 2048 ints
      __builtin_amdgcn_global_load_async_to_lds_b128(
          (gv4i_ptr)(gsrc + eidx), (lv4i_ptr)(mbuf + eidx),
          /*offset=*/0, /*cpol=*/0);
    }
  }
#else
  for (int i = tid; i < NS; i += 256) mbuf[i] = mask[(size_t)bidx * NS + i];
#endif

  // Zero the O reduction buffer while the mask is in flight.
  for (int i = tid; i < 16 * NH; i += 256) obuf[i] = 0.0f;

  // A = Q tile (16 x 64), loaded once per wave as two K=32 fragments.
  const float* qrow = q + ((size_t)bidx * NT + (t0 + ln16)) * NH;
  const v16h a0 = a_frag_from_f32(qrow, 0,  hi * 16);
  const v16h a1 = a_frag_from_f32(qrow, 32, hi * 16);

#if USE_ASYNC_LDS
  __builtin_amdgcn_s_wait_asynccnt(0);
#endif
  __syncthreads();   // mask + obuf visible to all waves

  // ---------------- Phase 1: masked scores via WMMA, fused row-max --------
  float lm[8];
#pragma unroll
  for (int r = 0; r < 8; ++r) lm[r] = -3.0e38f;

  // Wave w covers S columns [w*256, w*256+256) as 16 column-tiles of 16.
  for (int tl = 0; tl < 16; ++tl) {
    const int s0c = wave * 256 + tl * 16;
    // B = K^T chunk: lane's column = keys row (s0c + ln16), contiguous head dim.
    const float* krow = k + ((size_t)bidx * NS + (s0c + ln16)) * NH;
    const v16h b0 = b_frag_contig_f32(krow + 16 * hi);        // K rows 0..31
    const v16h b1 = b_frag_contig_f32(krow + 32 + 16 * hi);   // K rows 32..63
    v8f c = {};
    c = __builtin_amdgcn_wmma_f32_16x16x32_f16(false, a0, false, b0,
                                               (short)0, c, false, false);
    c = __builtin_amdgcn_wmma_f32_16x16x32_f16(false, a1, false, b1,
                                               (short)0, c, false, false);
    const int  mcol  = s0c + ln16;
    const bool valid = (mbuf[mcol] != 0);
#pragma unroll
    for (int r = 0; r < 8; ++r) {
      // C/D layout: VGPR r -> row r + 8*hi, lane -> column.
      const float val = valid ? c[r] : NEG_INF_F;
      lm[r] = fmaxf(lm[r], val);
      sc[(r + 8 * hi) * SCS + mcol] = val;
    }
  }
  // Butterfly row-max across the 16 lanes sharing each row half.
#pragma unroll
  for (int r = 0; r < 8; ++r) {
    float m = lm[r];
    m = fmaxf(m, __shfl_xor(m, 1));
    m = fmaxf(m, __shfl_xor(m, 2));
    m = fmaxf(m, __shfl_xor(m, 4));
    m = fmaxf(m, __shfl_xor(m, 8));
    lm[r] = m;
  }
  if (ln16 == 0) {
#pragma unroll
    for (int r = 0; r < 8; ++r) wmax[wave * 16 + 8 * hi + r] = lm[r];
  }
  __syncthreads();
  if (tid < 16) {
    float m = -3.0e38f;
#pragma unroll
    for (int w = 0; w < 8; ++w) m = fmaxf(m, wmax[w * 16 + tid]);
    rmax[tid] = m;
  }
  __syncthreads();

  // ---------------- Phase 2: exp + row-sum (conflict-free rows) -----------
  // Wave w owns rows {w, w+8}; lanes stride consecutive columns.
#pragma unroll
  for (int rsel = 0; rsel < 2; ++rsel) {
    const int   row = wave + 8 * rsel;
    const float rm  = rmax[row];
    float* srow = sc + row * SCS;
    float acc = 0.0f;
    for (int j = lane; j < NS; j += 32) {
      const float e = __expf(srow[j] - rm);
      srow[j] = e;
      acc += e;
    }
    acc += __shfl_xor(acc, 1);
    acc += __shfl_xor(acc, 2);
    acc += __shfl_xor(acc, 4);
    acc += __shfl_xor(acc, 8);
    acc += __shfl_xor(acc, 16);
    if (lane == 0) rsum[row] = acc;
  }
  __syncthreads();
  if (tid < 16) rsum[tid] = 1.0f / rsum[tid];   // in-place reciprocal
  __syncthreads();

  // ---------------- Phase 3a: finalize P, stream to HBM + f16 LDS ---------
  {
    float* pbase = p_out + ((size_t)bidx * NT + t0) * NS;
    for (int idx = tid; idx < 16 * NS; idx += 256) {   // coalesced columns
      const int   row = idx >> 11;
      const int   col = idx & (NS - 1);
      const float p   = sc[row * SCS + col] * rsum[row];
      pbase[idx]          = p;
      pb[row * PBS + col] = (_Float16)p;
    }
  }
  __syncthreads();

  // ---------------- Phase 3b: O = P V via WMMA ----------------
  // Wave w accumulates over its 256 S-columns: 8 chunks of K=32; output is
  // 16x64 = 4 N-tiles of 16.
  v8f acc[4] = {};
  for (int kc = 0; kc < 8; ++kc) {
    const int c0 = wave * 256 + kc * 32;
    // A fragment from f16 probabilities in LDS (two aligned 16B loads;
    // padded PBS stride rotates banks across lanes).
    const _Float16* pp = pb + (ln16 * PBS + c0 + 16 * hi);
    const v8h lo = *(const v8h*)(pp);
    const v8h hx = *(const v8h*)(pp + 8);
    const v16h af = __builtin_shufflevector(lo, hx, 0, 1, 2, 3, 4, 5, 6, 7,
                                            8, 9, 10, 11, 12, 13, 14, 15);
    // B fragment: V rows c0+16*hi+e, column nt*16+ln16 (L2-resident V).
    const float* vbase = v + ((size_t)bidx * NS + c0 + 16 * hi) * NH + ln16;
#pragma unroll
    for (int nt = 0; nt < 4; ++nt) {
      const v16h bf = b_frag_strided_f32(vbase + nt * 16, NH);
      acc[nt] = __builtin_amdgcn_wmma_f32_16x16x32_f16(false, af, false, bf,
                                                       (short)0, acc[nt],
                                                       false, false);
    }
  }
  // Cross-wave reduction of the 16x64 partials via LDS float atomics.
#pragma unroll
  for (int nt = 0; nt < 4; ++nt) {
#pragma unroll
    for (int r = 0; r < 8; ++r) {
      atomicAdd(&obuf[(r + 8 * hi) * NH + nt * 16 + ln16], acc[nt][r]);
    }
  }
  __syncthreads();

  // ---------------- Phase 3c: write O ----------------
  {
    float* obase = o_out + ((size_t)bidx * NT + t0) * NH;
    for (int idx = tid; idx < 16 * NH; idx += 256) obase[idx] = obuf[idx];
  }
}

extern "C" void kernel_launch(void* const* d_in, const int* in_sizes, int n_in,
                              void* d_out, int out_size, void* d_ws, size_t ws_size,
                              hipStream_t stream) {
  (void)in_sizes; (void)n_in; (void)d_ws; (void)ws_size; (void)out_size;
  const float* q    = (const float*)d_in[0];
  const float* k    = (const float*)d_in[1];
  const float* v    = (const float*)d_in[2];
  const int*   mask = (const int*)d_in[3];

  float* o_out = (float*)d_out;                       // (B,T,H) first
  float* p_out = o_out + (size_t)NB * NT * NH;        // then (B,T,S)

  const dim3 grid(NB * (NT / 16));  // 1024 workgroups
  const dim3 block(256);            // 8 waves (wave32)
  const size_t smem_bytes = 210048; // padded scores + f16 P + mask + reductions

  hipLaunchKernelGGL(attn_wmma_kernel, grid, block, smem_bytes, stream,
                     q, k, v, mask, o_out, p_out);
}
